// HDGNN_Influence_Model_86449101734281
// MI455X (gfx1250) — compile-verified, matrix operand order checked by reading, not run on hardware
//
#include <hip/hip_runtime.h>
#include <hip/hip_bf16.h>

// ---------------------------------------------------------------------------
// Problem constants (match reference)
// ---------------------------------------------------------------------------
constexpr int T = 10, N = 100000, G = 128, B = 256, R = 4, M = 32;
constexpr int E = 128, H = 2, D = 64, HID = 64;
constexpr int S = T * R * B;            // 10240 sequences

// ---------------------------------------------------------------------------
// CDNA5 WMMA helpers (wave32, 16x16x32 bf16 -> f32)
// ---------------------------------------------------------------------------
typedef __attribute__((ext_vector_type(16))) __bf16 v16bf;
typedef __attribute__((ext_vector_type(8)))  __bf16 v8bf;
typedef __attribute__((ext_vector_type(8)))  float  v8f;
typedef __attribute__((ext_vector_type(4)))  unsigned int v4u;
typedef __attribute__((ext_vector_type(8)))  int v8i;
typedef __attribute__((ext_vector_type(4)))  int v4i;

__device__ __forceinline__ __bf16 f2bf(float f) {
  unsigned u = __float_as_uint(f);
  u = (u + 0x7FFFu + ((u >> 16) & 1u)) >> 16;     // round-to-nearest-even
  unsigned short s = (unsigned short)u;
  __bf16 r;
  __builtin_memcpy(&r, &s, sizeof(r));
  return r;
}

__device__ __forceinline__ v8f wmma_bf16(v16bf a, v16bf b, v8f c) {
  // D = A(16x32 bf16) * B(32x16 bf16) + C(16x16 f32)
  return __builtin_amdgcn_wmma_f32_16x16x32_bf16(
      /*neg_a=*/false, a, /*neg_b=*/false, b,
      /*c_mod=*/(short)0, c, /*reuse_a=*/false, /*reuse_b=*/false);
}

// Load one 16x32 operand tile (A: rows of LHS, or B: W[n][k] for X*W^T) from a
// row-major bf16 array. Follows the ISA 16-bit 16x32 lane layout:
//   lane L (0..15):  row/col = base+L,    K = {k0..k0+7, k0+16..k0+23}
//   lane L (16..31): row/col = base+L-16, K = {k0+8..k0+15, k0+24..k0+31}
__device__ __forceinline__ v16bf load_tile(const __bf16* base, int stride,
                                           int rc0, int k0, int lane) {
  int rc = rc0 + (lane & 15);
  int kh = (lane >> 4) * 8;
  const __bf16* p = base + rc * stride + k0 + kh;     // 16B aligned
  v8bf lo = *(const v8bf*)(p);
  v8bf hi = *(const v8bf*)(p + 16);
  return __builtin_shufflevector(lo, hi, 0, 1, 2, 3, 4, 5, 6, 7,
                                 8, 9, 10, 11, 12, 13, 14, 15);
}

__device__ __forceinline__ float sigmoidf_(float x) {
  return 1.0f / (1.0f + __expf(-x));
}

// ---------------------------------------------------------------------------
// Tensor Data Mover: async DMA of a 2D bf16 panel (rows x cols, row-major,
// packed) from global memory into LDS. Builds the D# descriptor per
// CDNA5 ISA:
//   group0: count=1 | lds_addr | global_addr | type=2
//   group1: data_size=1 (2B), tensor_dim0/1 = cols/rows, tile_dim0/1 =
//           cols/rows (tile == tensor), tensor_dim0_stride = cols
// One instruction per issuing wave; completion tracked with TENSORcnt.
// This toolchain exposes the 6-arg builtin (..., int32x8 extra, i32 cpol).
// ---------------------------------------------------------------------------
__device__ __forceinline__ unsigned lds_off32(const void* p) {
  // On amdgcn the low 32 bits of a generic pointer to LDS are the LDS byte
  // offset (flat aperture maps addr[31:0] onto LDS).
  return (unsigned)(unsigned long long)(size_t)p;
}

__device__ __forceinline__ void tdm_load_2d_bf16(unsigned lds_addr,
                                                 const void* gptr,
                                                 unsigned rows,
                                                 unsigned cols) {
  unsigned long long ga = (unsigned long long)(size_t)gptr;
  v4u g0;
  g0.x = 1u;                                            // count=1, user mode
  g0.y = lds_addr;                                      // lds_addr (bytes)
  g0.z = (unsigned)(ga & 0xFFFFFFFFu);                  // global_addr[31:0]
  g0.w = (unsigned)((ga >> 32) & 0x1FFFFFFu) | (2u << 30); // addr[56:32]|type=2
  v8i g1;
  g1[0] = (int)(1u << 16);                              // data_size=1 (2 B)
  g1[1] = (int)((cols & 0xFFFFu) << 16);                // tensor_dim0 lo16
  g1[2] = (int)(((cols >> 16) & 0xFFFFu) |              // tensor_dim0 hi16
                ((rows & 0xFFFFu) << 16));              // tensor_dim1 lo16
  g1[3] = (int)(((rows >> 16) & 0xFFFFu) |              // tensor_dim1 hi16
                ((cols & 0xFFFFu) << 16));              // tile_dim0
  g1[4] = (int)(rows & 0xFFFFu);                        // tile_dim1 (dim2=0)
  g1[5] = (int)cols;                                    // dim0_stride[31:0]
  g1[6] = 0;                                            // stride hi / dim1_stride lo
  g1[7] = 0;
  v4i z4 = {0, 0, 0, 0};                                // groups 2/3 unused (2D)
  v8i z8 = {0, 0, 0, 0, 0, 0, 0, 0};
  __builtin_amdgcn_tensor_load_to_lds(g0, g1, z4, z4, z8, 0);
}

// ---------------------------------------------------------------------------
// Kernel 0: f32 -> bf16 weight conversion
// ---------------------------------------------------------------------------
__global__ void cvt_bf16_kernel(const float* __restrict__ in,
                                __bf16* __restrict__ out, int n) {
  int i = blockIdx.x * blockDim.x + threadIdx.x;
  if (i < n) out[i] = f2bf(in[i]);
}

// ---------------------------------------------------------------------------
// Kernel 1: neighbor gather + 2-layer MLP (128 -> relu 64 -> 32), WMMA
// One block (2 waves, 64 thr) per (t,r,b) sequence: X is [32 x 128].
// Weight panels arrive via TDM while the waves stage the gathered X.
// ---------------------------------------------------------------------------
__global__ void mlp_kernel(const float* __restrict__ emb,
                           const int* __restrict__ nidx,
                           const __bf16* __restrict__ w1,
                           const float* __restrict__ b1,
                           const __bf16* __restrict__ w2,
                           const float* __restrict__ b2,
                           __bf16* __restrict__ xout) {
  __shared__ __align__(16) __bf16 Xs[32][128];
  __shared__ __align__(16) __bf16 W1s[64 * 128];
  __shared__ __align__(16) __bf16 W2s[32 * 64];
  __shared__ __align__(16) __bf16 H1s[32][64];

  const int seq = blockIdx.x;                 // [T][R][B] order
  const int t   = seq / (R * B);
  const int tid = threadIdx.x;
  const int* nb = nidx + (size_t)seq * M;

  if (tid < 32) {                             // wave 0 drives the TDM
    tdm_load_2d_bf16(lds_off32(W1s), w1, 64, 128);
    tdm_load_2d_bf16(lds_off32(W2s), w2, 32, 64);
  }
  for (int i = tid; i < 32 * 128; i += 64) {  // meanwhile: gather + cvt X
    int row = i >> 7, col = i & 127;
    int nn = nb[row];
    Xs[row][col] = f2bf(emb[((size_t)t * N + (size_t)nn) * G + col]);
  }
  if (tid < 32) __builtin_amdgcn_s_wait_tensorcnt(0);
  __syncthreads();

  const int lane  = tid & 31;
  const int m0    = (tid >> 5) * 16;
  const int ccol  = lane & 15;
  const int crow0 = m0 + (lane >> 4) * 8;

  // stage 1: H1 = relu(X * W1^T + b1)   [32x64]
  for (int nt = 0; nt < 4; ++nt) {
    v8f acc = {};
#pragma unroll
    for (int kc = 0; kc < 4; ++kc) {
      v16bf a  = load_tile(&Xs[0][0], 128, m0, kc * 32, lane);
      v16bf bm = load_tile(W1s, 128, nt * 16, kc * 32, lane);
      acc = wmma_bf16(a, bm, acc);
    }
    int col = nt * 16 + ccol;
    float bias = b1[col];
#pragma unroll
    for (int j = 0; j < 8; ++j) {
      float hv = acc[j] + bias;
      H1s[crow0 + j][col] = f2bf(hv > 0.0f ? hv : 0.0f);
    }
  }
  __syncthreads();

  // stage 2: out = H1 * W2^T + b2   [32x32] -> bf16 GRU input
  for (int nt = 0; nt < 2; ++nt) {
    v8f acc = {};
#pragma unroll
    for (int kc = 0; kc < 2; ++kc) {
      v16bf a  = load_tile(&H1s[0][0], 64, m0, kc * 32, lane);
      v16bf bm = load_tile(W2s, 64, nt * 16, kc * 32, lane);
      acc = wmma_bf16(a, bm, acc);
    }
    int col = nt * 16 + ccol;
    float bias = b2[col];
#pragma unroll
    for (int j = 0; j < 8; ++j)
      xout[((size_t)seq * 32 + (crow0 + j)) * 32 + col] = f2bf(acc[j] + bias);
  }
}

// ---------------------------------------------------------------------------
// Kernel 2: GRU (one wave = 16 sequences in the WMMA M dimension)
// blockIdx.y: 0 = forward, 1 = backward (reads time-reversed inputs).
// Gate weights arrive via TDM; per-step gate GEMMs are 16x16x32 bf16 WMMA.
// Writes final hidden state into value[t][1+r][b][dirOff + 0..63].
// ---------------------------------------------------------------------------
__global__ void gru_kernel(const __bf16* __restrict__ xgru,
                           const int* __restrict__ nlen,
                           const __bf16* __restrict__ wih_f,
                           const __bf16* __restrict__ whh_f,
                           const float* __restrict__ bih_f,
                           const float* __restrict__ bhh_f,
                           const __bf16* __restrict__ wih_b,
                           const __bf16* __restrict__ whh_b,
                           const float* __restrict__ bih_b,
                           const float* __restrict__ bhh_b,
                           float* __restrict__ value) {
  __shared__ __align__(16) __bf16 Wih[192 * 32];
  __shared__ __align__(16) __bf16 Whh[192 * 64];
  __shared__ float Bih[192], Bhh[192];
  __shared__ __align__(16) float  Hbuf[2][16][64];
  __shared__ __align__(16) __bf16 Hbf[16][64];
  __shared__ __align__(16) __bf16 Xs[16][32];
  __shared__ int Lens[16];

  const int lane = threadIdx.x;
  const int dir  = blockIdx.y;               // 0 fwd, 1 bwd
  const int seqbase = blockIdx.x * 16;

  const __bf16* wih = dir ? wih_b : wih_f;
  const __bf16* whh = dir ? whh_b : whh_f;
  const float* bihp = dir ? bih_b : bih_f;
  const float* bhhp = dir ? bhh_b : bhh_f;

  // async weight DMA (single wave -> one TDM op per wave, EXEC ignored)
  tdm_load_2d_bf16(lds_off32(Wih), wih, 192, 32);
  tdm_load_2d_bf16(lds_off32(Whh), whh, 192, 64);

  for (int i = lane; i < 192; i += 32) { Bih[i] = bihp[i]; Bhh[i] = bhhp[i]; }
  if (lane < 16) Lens[lane] = nlen[seqbase + lane];
  for (int i = lane; i < 16 * 64; i += 32) {
    (&Hbuf[0][0][0])[i] = 0.0f;
    (&Hbf[0][0])[i] = f2bf(0.0f);
  }
  __builtin_amdgcn_s_wait_tensorcnt(0);
  __syncthreads();

  const int ccol  = lane & 15;
  const int crow0 = (lane >> 4) * 8;

  for (int step = 0; step < M; ++step) {
    const int oldIdx = step & 1;
    // stage x_t (time-reversed + length-clamped for backward dir)
    for (int i = lane; i < 16 * 32; i += 32) {
      int row = i >> 5, col = i & 31;
      int m = step;
      if (dir) {
        m = Lens[row] - 1 - step;
        m = m < 0 ? 0 : (m > 31 ? 31 : m);
      }
      Xs[row][col] = xgru[(((size_t)(seqbase + row)) * 32 + m) * 32 + col];
    }
    __syncthreads();

    for (int ht = 0; ht < 4; ++ht) {          // 4 x 16 columns of HID
      const int n0 = ht * 16;
      v8f ir = {}, iz = {}, in_ = {}, hr = {}, hz = {}, hn = {};
      v16bf ax = load_tile(&Xs[0][0], 32, 0, 0, lane);
      ir  = wmma_bf16(ax, load_tile(Wih, 32,       n0, 0, lane), ir);
      iz  = wmma_bf16(ax, load_tile(Wih, 32,  64 + n0, 0, lane), iz);
      in_ = wmma_bf16(ax, load_tile(Wih, 32, 128 + n0, 0, lane), in_);
#pragma unroll
      for (int kc = 0; kc < 2; ++kc) {
        v16bf ah = load_tile(&Hbf[0][0], 64, 0, kc * 32, lane);
        hr = wmma_bf16(ah, load_tile(Whh, 64,       n0, kc * 32, lane), hr);
        hz = wmma_bf16(ah, load_tile(Whh, 64,  64 + n0, kc * 32, lane), hz);
        hn = wmma_bf16(ah, load_tile(Whh, 64, 128 + n0, kc * 32, lane), hn);
      }
      const int col = n0 + ccol;
      const float bir = Bih[col], biz = Bih[64 + col], bin_ = Bih[128 + col];
      const float bhr = Bhh[col], bhz = Bhh[64 + col], bhn  = Bhh[128 + col];
#pragma unroll
      for (int j = 0; j < 8; ++j) {
        int row = crow0 + j;
        float hv = Hbuf[oldIdx][row][col];
        float r = sigmoidf_(ir[j] + bir + hr[j] + bhr);
        float z = sigmoidf_(iz[j] + biz + hz[j] + bhz);
        float n = tanhf(in_[j] + bin_ + r * (hn[j] + bhn));
        float hnew = (1.0f - z) * n + z * hv;
        Hbuf[1 - oldIdx][row][col] = (step < Lens[row]) ? hnew : hv;
      }
    }
    __syncthreads();
    for (int i = lane; i < 16 * 64; i += 32)
      (&Hbf[0][0])[i] = f2bf((&Hbuf[1 - oldIdx][0][0])[i]);
    __syncthreads();
  }

  // after 32 steps final state sits in Hbuf[0]
  const int dirOff = dir ? 0 : 64;            // feat = concat([h_bwd, h_fwd])
  for (int i = lane; i < 16 * 64; i += 32) {
    int row = i >> 6, col = i & 63;
    int s = seqbase + row;
    int t = s / (R * B);
    int rr = (s / B) % R;
    int b = s % B;
    value[(((size_t)t * 5 + (1 + rr)) * B + b) * E + dirOff + col] =
        Hbuf[0][row][col];
  }
}

// ---------------------------------------------------------------------------
// Kernel 3: self-features -> value[t][0][b][:]
// ---------------------------------------------------------------------------
__global__ void selffeat_kernel(const float* __restrict__ emb,
                                const int* __restrict__ input_ids,
                                const int* __restrict__ align_list,
                                float* __restrict__ value) {
  int i = blockIdx.x * blockDim.x + threadIdx.x;      // T*B*E
  if (i >= T * B * E) return;
  int e = i % E;
  int b = (i / E) % B;
  int t = i / (B * E);
  int id = input_ids[b];
  int idx = align_list[(size_t)id * T + t];
  if (idx < 0) idx = 0;
  float val = (idx == 0) ? 0.0f                        // row 0 zeroed
                         : emb[((size_t)t * N + idx) * G + e];
  value[(((size_t)t * 5 + 0) * B + b) * E + e] = val;
}

// ---------------------------------------------------------------------------
// Kernel 4: generic 128x128 linear (out = in * W^T + bias), WMMA
// block = 128 thr (4 waves), 64 rows per block; rows % 64 == 0 for all calls.
// Weight panel arrives via TDM while all waves stage/convert the activations.
// ---------------------------------------------------------------------------
__global__ void linear128_kernel(const float* __restrict__ in,
                                 const __bf16* __restrict__ w,
                                 const float* __restrict__ bias,
                                 float* __restrict__ out) {
  __shared__ __align__(16) __bf16 As[64 * 128];
  __shared__ __align__(16) __bf16 Ws[128 * 128];
  __shared__ float Bs[128];
  const int tid = threadIdx.x;
  const size_t row0 = (size_t)blockIdx.x * 64;

  if (tid < 32) tdm_load_2d_bf16(lds_off32(Ws), w, 128, 128);
  for (int i = tid; i < 64 * 128; i += 128) As[i] = f2bf(in[row0 * 128 + i]);
  if (tid < 128) Bs[tid] = bias[tid];
  if (tid < 32) __builtin_amdgcn_s_wait_tensorcnt(0);
  __syncthreads();

  const int lane  = tid & 31;
  const int m0    = (tid >> 5) * 16;
  const int ccol  = lane & 15;
  const int crow0 = m0 + (lane >> 4) * 8;

  for (int nt = 0; nt < 8; ++nt) {
    v8f acc = {};
#pragma unroll
    for (int kc = 0; kc < 4; ++kc) {
      v16bf a  = load_tile(As, 128, m0, kc * 32, lane);
      v16bf bm = load_tile(Ws, 128, nt * 16, kc * 32, lane);
      acc = wmma_bf16(a, bm, acc);
    }
    int col = nt * 16 + ccol;
    float bb = Bs[col];
#pragma unroll
    for (int j = 0; j < 8; ++j)
      out[(row0 + crow0 + j) * 128 + col] = acc[j] + bb;
  }
}

// ---------------------------------------------------------------------------
// Kernel 5: tiny attention (5 keys per query, softmax in registers)
// one thread per (t, b, h, qi)
// ---------------------------------------------------------------------------
__global__ void attn_kernel(const float* __restrict__ q,
                            const float* __restrict__ k,
                            const float* __restrict__ v,
                            float* __restrict__ ctx) {
  int i = blockIdx.x * blockDim.x + threadIdx.x;       // T*B*H*5 = 25600
  if (i >= T * B * H * 5) return;
  int qi = i % 5;
  int h  = (i / 5) % H;
  int b  = (i / (5 * H)) % B;
  int t  = i / (5 * H * B);

  const float* qp = q + ((size_t)qi * B + b) * E + h * D;
  float sc[5];
  float mx = -1e30f;
#pragma unroll
  for (int kk = 0; kk < 5; ++kk) {
    const float* kp = k + (((size_t)t * 5 + kk) * B + b) * E + h * D;
    float s = 0.0f;
    for (int d = 0; d < D; ++d) s += qp[d] * kp[d];
    sc[kk] = s * 0.125f;                                // 1/sqrt(64)
    mx = fmaxf(mx, sc[kk]);
  }
  float sum = 0.0f;
#pragma unroll
  for (int kk = 0; kk < 5; ++kk) { sc[kk] = __expf(sc[kk] - mx); sum += sc[kk]; }
  float inv = 1.0f / sum;
  float* cp = ctx + (((size_t)t * 5 + qi) * B + b) * E + h * D;
  for (int d = 0; d < D; ++d) {
    float acc = 0.0f;
#pragma unroll
    for (int kk = 0; kk < 5; ++kk)
      acc += sc[kk] * v[(((size_t)t * 5 + kk) * B + b) * E + h * D + d];
    cp[d] = acc * inv;
  }
}

// ---------------------------------------------------------------------------
// Kernel 6: mean over the 5 query slots -> d_out ([T][B][E] flat)
// ---------------------------------------------------------------------------
__global__ void mean5_kernel(const float* __restrict__ proj,
                             float* __restrict__ out) {
  int i = blockIdx.x * blockDim.x + threadIdx.x;       // T*B*E
  if (i >= T * B * E) return;
  int e = i % E;
  int b = (i / E) % B;
  int t = i / (B * E);
  float acc = 0.0f;
#pragma unroll
  for (int qi = 0; qi < 5; ++qi)
    acc += proj[(((size_t)t * 5 + qi) * B + b) * E + e];
  out[i] = acc * 0.2f;
}

// ---------------------------------------------------------------------------
// Host driver
// ---------------------------------------------------------------------------
extern "C" void kernel_launch(void* const* d_in, const int* in_sizes, int n_in,
                              void* d_out, int out_size, void* d_ws,
                              size_t ws_size, hipStream_t stream) {
  (void)in_sizes; (void)n_in; (void)out_size; (void)ws_size;

  const float* emb       = (const float*)d_in[0];
  const float* mlp_w1    = (const float*)d_in[1];
  const float* mlp_b1    = (const float*)d_in[2];
  const float* mlp_w2    = (const float*)d_in[3];
  const float* mlp_b2    = (const float*)d_in[4];
  const float* g_wih_f   = (const float*)d_in[5];
  const float* g_whh_f   = (const float*)d_in[6];
  const float* g_bih_f   = (const float*)d_in[7];
  const float* g_bhh_f   = (const float*)d_in[8];
  const float* g_wih_b   = (const float*)d_in[9];
  const float* g_whh_b   = (const float*)d_in[10];
  const float* g_bih_b   = (const float*)d_in[11];
  const float* g_bhh_b   = (const float*)d_in[12];
  const float* attn_in_w = (const float*)d_in[13];
  const float* attn_in_b = (const float*)d_in[14];
  const float* attn_o_w  = (const float*)d_in[15];
  const float* attn_o_b  = (const float*)d_in[16];
  const float* query     = (const float*)d_in[17];
  const int*   input_ids = (const int*)d_in[18];
  const int*   align_lst = (const int*)d_in[19];
  const int*   nidx      = (const int*)d_in[20];
  const int*   nlen      = (const int*)d_in[21];

  char* ws = (char*)d_ws;
  size_t off = 0;
  auto alloc = [&](size_t bytes) -> void* {
    off = (off + 255) & ~(size_t)255;
    void* p = ws + off;
    off += bytes;
    return p;
  };

  __bf16* w1_bf   = (__bf16*)alloc(64 * 128 * 2);
  __bf16* w2_bf   = (__bf16*)alloc(32 * 64 * 2);
  __bf16* wihf_bf = (__bf16*)alloc(192 * 32 * 2);
  __bf16* whhf_bf = (__bf16*)alloc(192 * 64 * 2);
  __bf16* wihb_bf = (__bf16*)alloc(192 * 32 * 2);
  __bf16* whhb_bf = (__bf16*)alloc(192 * 64 * 2);
  __bf16* wqkv_bf = (__bf16*)alloc(384 * 128 * 2);
  __bf16* wout_bf = (__bf16*)alloc(128 * 128 * 2);
  __bf16* xgru    = (__bf16*)alloc((size_t)S * M * 32 * 2);     // ~21 MB
  float*  value   = (float*)alloc((size_t)T * 5 * B * E * 4);   // 6.5 MB
  float*  qbuf    = (float*)alloc((size_t)5 * B * E * 4);
  float*  kbuf    = (float*)alloc((size_t)T * 5 * B * E * 4);
  float*  vbuf    = (float*)alloc((size_t)T * 5 * B * E * 4);
  float*  ctx     = (float*)alloc((size_t)T * 5 * B * E * 4);
  float*  proj    = (float*)alloc((size_t)T * 5 * B * E * 4);

  auto cvt = [&](const float* src, __bf16* dst, int n) {
    cvt_bf16_kernel<<<(n + 255) / 256, 256, 0, stream>>>(src, dst, n);
  };
  cvt(mlp_w1, w1_bf, 64 * 128);
  cvt(mlp_w2, w2_bf, 32 * 64);
  cvt(g_wih_f, wihf_bf, 192 * 32);
  cvt(g_whh_f, whhf_bf, 192 * 64);
  cvt(g_wih_b, wihb_bf, 192 * 32);
  cvt(g_whh_b, whhb_bf, 192 * 64);
  cvt(attn_in_w, wqkv_bf, 384 * 128);
  cvt(attn_o_w, wout_bf, 128 * 128);

  // 1) gather + MLP (WMMA, TDM weight staging)
  mlp_kernel<<<S, 64, 0, stream>>>(emb, nidx, w1_bf, mlp_b1, w2_bf, mlp_b2,
                                   xgru);
  // 2) bidirectional GRU (WMMA per step, TDM weight staging)
  gru_kernel<<<dim3(S / 16, 2), 32, 0, stream>>>(
      xgru, nlen, wihf_bf, whhf_bf, g_bih_f, g_bhh_f, wihb_bf, whhb_bf,
      g_bih_b, g_bhh_b, value);
  // 3) self features into value slot 0
  selffeat_kernel<<<(T * B * E + 255) / 256, 256, 0, stream>>>(
      emb, input_ids, align_lst, value);
  // 4) Q/K/V projections (WMMA)
  linear128_kernel<<<(5 * B) / 64, 128, 0, stream>>>(query, wqkv_bf,
                                                     attn_in_b, qbuf);
  linear128_kernel<<<(T * 5 * B) / 64, 128, 0, stream>>>(
      value, wqkv_bf + 128 * 128, attn_in_b + 128, kbuf);
  linear128_kernel<<<(T * 5 * B) / 64, 128, 0, stream>>>(
      value, wqkv_bf + 2 * 128 * 128, attn_in_b + 256, vbuf);
  // 5) attention (tiny 5x5 softmax per (t,b,h,q))
  attn_kernel<<<(T * B * H * 5 + 127) / 128, 128, 0, stream>>>(qbuf, kbuf,
                                                               vbuf, ctx);
  // 6) output projection (WMMA) + mean over the 5 query slots
  linear128_kernel<<<(T * 5 * B) / 64, 128, 0, stream>>>(ctx, wout_bf,
                                                         attn_o_b, proj);
  mean5_kernel<<<(T * B * E + 255) / 256, 256, 0, stream>>>(proj,
                                                            (float*)d_out);
}
